// Encoder2d_61108794687738
// MI455X (gfx1250) — matrix operand order repaired and checked by reading, not compiled, not run on hardware
//
#include <hip/hip_runtime.h>
#include <hip/hip_bf16.h>
#include <math.h>

// ---------------- WMMA types / helpers (gfx1250, wave32) ----------------
typedef _Float16 h16v __attribute__((ext_vector_type(16)));
typedef _Float16 h2v  __attribute__((ext_vector_type(2)));
typedef float    f8v  __attribute__((ext_vector_type(8)));

__device__ __forceinline__ f8v wmma_f16(h16v a, h16v b, f8v c) {
  // D = A(16x32 f16) * B(32x16 f16) + C(16x16 f32)
  return __builtin_amdgcn_wmma_f32_16x16x32_f16(
      /*neg_a=*/false, a, /*neg_b=*/false, b,
      /*c_mod=*/(short)0, c, /*reuse_a=*/false, /*reuse_b=*/false);
}

// A fragment: 16x32 f16 tile, row-major source. lane&15 = M row, lane>>4 = K half.
// ISA layout: VGPR j (pair 2j,2j+1): K = (j<4?0:16) + 8*half + 2*(j&3) (+0/+1)
__device__ __forceinline__ h16v load_A(const _Float16* row, int kb, int half) {
  h16v a;
#pragma unroll
  for (int j = 0; j < 8; ++j) {
    int k0 = kb + ((j < 4) ? 0 : 16) + 8 * half + 2 * (j & 3);
    h2v p = *(const h2v*)(row + k0);
    a[2 * j] = p[0];
    a[2 * j + 1] = p[1];
  }
  return a;
}

// B fragments are pre-swizzled: tile (tk,tn): 32 lanes x 16 f16 contiguous.
__device__ __forceinline__ h16v load_B(const _Float16* Bp, int tk, int tn,
                                       int TN, int lane) {
  return *(const h16v*)(Bp + (((size_t)(tk * TN + tn) * 32 + lane) * 16));
}

__device__ __forceinline__ float leakyf(float v) { return v >= 0.f ? v : 0.01f * v; }
__device__ __forceinline__ float sigf(float v) { return 1.f / (1.f + expf(-v)); }

// ---------------- weight pack: B-fragment swizzle (f32 -> f16) ----------------
// lane L holds column n = tn*16 + (L&15); element e: K = tk*32 + (e<8 ? 8*(L>>4) + e : 16 + 8*(L>>4) + e - 8)
__global__ void k_pack(const float* __restrict__ W, _Float16* __restrict__ out,
                       int K, int Nc, int Kpad, int trans) {
  int idx = blockIdx.x * blockDim.x + threadIdx.x;
  int TN = Nc / 16;
  int total = (Kpad / 32) * TN * 512;
  if (idx >= total) return;
  int tile = idx >> 9, within = idx & 511, lane = within >> 4, e = within & 15;
  int tk = tile / TN, tn = tile % TN;
  int n = tn * 16 + (lane & 15);
  int k = tk * 32 + ((e < 8) ? 8 * (lane >> 4) + e : 16 + 8 * (lane >> 4) + (e - 8));
  float v = 0.f;
  if (k < K) v = trans ? W[(size_t)n * K + k] : W[(size_t)k * Nc + n];
  out[idx] = (_Float16)v;
}

__global__ void k_cvt(const float* __restrict__ in, _Float16* __restrict__ out,
                      size_t n) {
  size_t i = (size_t)blockIdx.x * blockDim.x + threadIdx.x;
  if (i < n) out[i] = (_Float16)in[i];
}

__global__ void k_zero(float* __restrict__ p, size_t n) {
  size_t i = (size_t)blockIdx.x * blockDim.x + threadIdx.x;
  if (i < n) p[i] = 0.f;
}

// ---------------- generic WMMA GEMM ----------------
// C[M,Nc] = act(A16[M,K] @ Bp + bias). K % 32 == 0, (Nc/16) % NT == 0.
// One wave computes NT adjacent 16x16 tiles: A fragment loaded once per K-chunk.
template <int NT>
__global__ __launch_bounds__(256) void k_gemm(
    const _Float16* __restrict__ A, int lda, const _Float16* __restrict__ Bp,
    int K, int Nc, const float* __restrict__ bias, int act,
    float* __restrict__ C, _Float16* __restrict__ C16, int M) {
  int TN = Nc / 16;
  int groups = TN / NT;
  int wid = blockIdx.x * (blockDim.x >> 5) + (threadIdx.x >> 5);
  int rowTiles = (M + 15) >> 4;
  if (wid >= rowTiles * groups) return;
  int g = wid % groups, tr = wid / groups;
  int tn0 = g * NT;
  int lane = threadIdx.x & 31, half = lane >> 4, mr = lane & 15;
  int rowBase = tr << 4;
  int arow = rowBase + mr;
  if (arow >= M) arow = M - 1;
  const _Float16* rowPtr = A + (size_t)arow * lda;
  f8v acc[NT] = {};
  for (int kb = 0; kb < K; kb += 32) {
    h16v a = load_A(rowPtr, kb, half);
#pragma unroll
    for (int t = 0; t < NT; ++t)
      acc[t] = wmma_f16(a, load_B(Bp, kb >> 5, tn0 + t, TN, lane), acc[t]);
  }
#pragma unroll
  for (int t = 0; t < NT; ++t) {
    int col = (tn0 + t) * 16 + mr;
    float bv = bias ? bias[col] : 0.f;
#pragma unroll
    for (int r = 0; r < 8; ++r) {
      int m = rowBase + 8 * half + r;
      if (m < M) {
        float v = acc[t][r] + bv;
        if (act == 1) v = leakyf(v);
        C[(size_t)m * Nc + col] = v;
        if (C16) C16[(size_t)m * Nc + col] = (_Float16)v;
      }
    }
  }
}

// ---------------- edge layer 0 ----------------
// concat tail chunk: K 96..111 from edge_attr, K 112..127 zero pad
__device__ __forceinline__ h16v load_A_concat_tail(const _Float16* earow, int half) {
  h16v a;
#pragma unroll
  for (int j = 0; j < 8; ++j) {
    if (j < 4) {
      int k0 = 8 * half + 2 * j;
      h2v p = *(const h2v*)(earow + k0);
      a[2 * j] = p[0];
      a[2 * j + 1] = p[1];
    } else {
      a[2 * j] = (_Float16)0.f;
      a[2 * j + 1] = (_Float16)0.f;
    }
  }
  return a;
}

// compute all 6 col-tiles of mj = concat(x16[src], ea16) @ e_lin1 for a 16-edge tile
__device__ __forceinline__ void edge_mj_gemm(const _Float16* xrow,
                                             const _Float16* earow,
                                             const _Float16* pE1, int lane,
                                             int half, f8v (&acc)[6]) {
  for (int kb = 0; kb < 96; kb += 32) {
    h16v a = load_A(xrow, kb, half);
#pragma unroll
    for (int t = 0; t < 6; ++t)
      acc[t] = wmma_f16(a, load_B(pE1, kb >> 5, t, 6, lane), acc[t]);
  }
  h16v a = load_A_concat_tail(earow, half);
#pragma unroll
  for (int t = 0; t < 6; ++t)
    acc[t] = wmma_f16(a, load_B(pE1, 3, t, 6, lane), acc[t]);
}

// logits pass: logit = leaky(mj.att_l + x[dst].att_r); e=exp(logit); s_sum[dst]+=e
__global__ __launch_bounds__(256) void k_edge_logits(
    const _Float16* __restrict__ x16, const _Float16* __restrict__ ea16,
    const long long* __restrict__ src, const long long* __restrict__ dst,
    const _Float16* __restrict__ pE1, const float* __restrict__ att_l,
    const float* __restrict__ att_r, const float* __restrict__ x32,
    float* __restrict__ e_edge, float* __restrict__ s_sum, int E) {
  __shared__ float slog[8][16];
  int w = threadIdx.x >> 5, lane = threadIdx.x & 31;
  int half = lane >> 4, mr = lane & 15;
  int tile = blockIdx.x * 8 + w;
  int nt = (E + 15) >> 4;
  bool active = tile < nt;
  if (lane < 16) slog[w][lane] = 0.f;
  __syncthreads();
  if (active) {
    int edge = tile * 16 + mr;
    if (edge >= E) edge = E - 1;
    long long s = src[edge];
    f8v acc[6] = {};
    edge_mj_gemm(x16 + (size_t)s * 96, ea16 + (size_t)edge * 16, pE1, lane,
                 half, acc);
#pragma unroll
    for (int t = 0; t < 6; ++t) {
      float al = att_l[t * 16 + mr];
#pragma unroll
      for (int r = 0; r < 8; ++r)
        atomicAdd(&slog[w][8 * half + r], leakyf(acc[t][r]) * al);
    }
  }
  __syncthreads();
  if (active && lane < 16) {
    int edge = tile * 16 + lane;
    if (edge < E) {
      long long d = dst[edge];
      const float4* xr = (const float4*)(x32 + (size_t)d * 96);
      const float4* ar = (const float4*)att_r;
      float dv = 0.f;
#pragma unroll 4
      for (int c = 0; c < 24; ++c) {
        float4 xv = xr[c], av = ar[c];
        dv += xv.x * av.x + xv.y * av.y + xv.z * av.z + xv.w * av.w;
      }
      float lg = leakyf(slog[w][lane] + dv);
      float ev = expf(lg);
      e_edge[edge] = ev;
      atomicAdd(&s_sum[d], ev);
    }
  }
}

// message pass: recompute mj (LDS stage), msg = (mj @ e_lin2)*alpha, scatter
__global__ __launch_bounds__(256) void k_edge_msg(
    const _Float16* __restrict__ x16, const _Float16* __restrict__ ea16,
    const long long* __restrict__ src, const long long* __restrict__ dst,
    const _Float16* __restrict__ pE1, const _Float16* __restrict__ pE2,
    const float* __restrict__ e_edge, const float* __restrict__ s_sum,
    float* __restrict__ h32, int E) {
  __shared__ _Float16 mjL[8][16][96];
  __shared__ float wts[8][16];
  __shared__ int dl[8][16];
  int w = threadIdx.x >> 5, lane = threadIdx.x & 31;
  int half = lane >> 4, mr = lane & 15;
  int tile = blockIdx.x * 8 + w;
  int nt = (E + 15) >> 4;
  if (tile >= nt) return;
  int edge = tile * 16 + mr;
  if (edge >= E) edge = E - 1;
  if (lane < 16) {
    long long d = dst[edge];
    dl[w][lane] = (int)d;
    wts[w][lane] =
        (tile * 16 + lane < E) ? e_edge[edge] / (s_sum[d] + 1e-16f) : 0.f;
  }
  long long s = src[edge];
  // pass 1: mj tile -> LDS (D-frag layout converted to row-major f16)
  {
    f8v acc[6] = {};
    edge_mj_gemm(x16 + (size_t)s * 96, ea16 + (size_t)edge * 16, pE1, lane,
                 half, acc);
#pragma unroll
    for (int t = 0; t < 6; ++t)
#pragma unroll
      for (int r = 0; r < 8; ++r)
        mjL[w][8 * half + r][t * 16 + mr] = (_Float16)leakyf(acc[t][r]);
  }
  // pass 2: msg = mj @ e_lin2 (LDS ops are in-order per wave), scale, scatter
  {
    f8v acc[6] = {};
    for (int kb = 0; kb < 96; kb += 32) {
      h16v a = load_A(&mjL[w][mr][0], kb, half);
#pragma unroll
      for (int t = 0; t < 6; ++t)
        acc[t] = wmma_f16(a, load_B(pE2, kb >> 5, t, 6, lane), acc[t]);
    }
#pragma unroll
    for (int t = 0; t < 6; ++t) {
      int col = t * 16 + mr;
#pragma unroll
      for (int r = 0; r < 8; ++r) {
        int m = 8 * half + r;
        if (tile * 16 + m < E)
          atomicAdd(&h32[(size_t)dl[w][m] * 96 + col], acc[t][r] * wts[w][m]);
      }
    }
  }
}

// ---------------- GAT helpers ----------------
__global__ void k_node_att(const float* __restrict__ xw,
                           const float* __restrict__ wsrc,
                           const float* __restrict__ wdst,
                           float* __restrict__ a_src, float* __restrict__ a_dst,
                           int Nn) {
  int i = blockIdx.x * blockDim.x + threadIdx.x;
  if (i >= Nn) return;
  const float4* r = (const float4*)(xw + (size_t)i * 96);
  const float4* s4 = (const float4*)wsrc;
  const float4* d4 = (const float4*)wdst;
  float s1 = 0.f, s2 = 0.f;
#pragma unroll 4
  for (int c = 0; c < 24; ++c) {
    float4 xv = r[c], sv = s4[c], dv = d4[c];
    s1 += xv.x * sv.x + xv.y * sv.y + xv.z * sv.z + xv.w * sv.w;
    s2 += xv.x * dv.x + xv.y * dv.y + xv.z * dv.z + xv.w * dv.w;
  }
  a_src[i] = s1;
  a_dst[i] = s2;
}

__global__ void k_gat_logits(const float* __restrict__ a_src,
                             const float* __restrict__ a_dst,
                             const long long* __restrict__ src,
                             const long long* __restrict__ dst,
                             float* __restrict__ e_edge,
                             float* __restrict__ s_sum, int E) {
  int e = blockIdx.x * blockDim.x + threadIdx.x;
  if (e >= E) return;
  long long d = dst[e];
  float lg = leakyf(a_src[src[e]] + a_dst[d]);
  float ev = expf(lg);
  e_edge[e] = ev;
  atomicAdd(&s_sum[d], ev);
}

__global__ void k_gat_scatter(const float* __restrict__ xw,
                              const long long* __restrict__ src,
                              const long long* __restrict__ dst,
                              const float* __restrict__ e_edge,
                              const float* __restrict__ s_sum,
                              float* __restrict__ h32, int E) {
  long long idx = (long long)blockIdx.x * blockDim.x + threadIdx.x;
  long long e = idx / 24;
  int g = (int)(idx % 24);
  if (e >= E) return;
  long long d = dst[e], s = src[e];
  float wv = e_edge[e] / (s_sum[d] + 1e-16f);
  float4 xv = *(const float4*)(xw + (size_t)s * 96 + 4 * g);
  float* hr = h32 + (size_t)d * 96 + 4 * g;
  atomicAdd(&hr[0], xv.x * wv);
  atomicAdd(&hr[1], xv.y * wv);
  atomicAdd(&hr[2], xv.z * wv);
  atomicAdd(&hr[3], xv.w * wv);
}

// ---------------- elementwise ----------------
__global__ void k_elu_bias(float* __restrict__ h32,
                           const float* __restrict__ bias,
                           _Float16* __restrict__ h16, size_t total) {
  size_t i = (size_t)blockIdx.x * blockDim.x + threadIdx.x;
  if (i >= total) return;
  int c = (int)(i % 96);
  float v = h32[i] + bias[c];
  v = v > 0.f ? v : expm1f(v);
  h32[i] = v;
  h16[i] = (_Float16)v;
}

__global__ void k_gru_combine(const float* __restrict__ gi,
                              const float* __restrict__ gh,
                              float* __restrict__ x32,
                              _Float16* __restrict__ x16, size_t total) {
  size_t i = (size_t)blockIdx.x * blockDim.x + threadIdx.x;
  if (i >= total) return;
  size_t node = i / 96;
  int c = (int)(i % 96);
  size_t base = node * 288;
  float r = sigf(gi[base + c] + gh[base + c]);
  float z = sigf(gi[base + 96 + c] + gh[base + 96 + c]);
  float n = tanhf(gi[base + 192 + c] + r * gh[base + 192 + c]);
  float hid = x32[i];
  float v = (1.f - z) * n + z * hid;
  v = fmaxf(v, 0.f);
  x32[i] = v;
  x16[i] = (_Float16)v;
}

__global__ void k_layernorm(const float* __restrict__ x32,
                            float* __restrict__ out, int Nn) {
  int i = blockIdx.x * blockDim.x + threadIdx.x;
  if (i >= Nn) return;
  const float4* r4 = (const float4*)(x32 + (size_t)i * 96);
  float4 buf[24];
  float mu = 0.f;
#pragma unroll
  for (int c = 0; c < 24; ++c) {
    buf[c] = r4[c];
    mu += buf[c].x + buf[c].y + buf[c].z + buf[c].w;
  }
  mu *= (1.f / 96.f);
  float var = 0.f;
#pragma unroll
  for (int c = 0; c < 24; ++c) {
    float dx = buf[c].x - mu, dy = buf[c].y - mu;
    float dz = buf[c].z - mu, dw = buf[c].w - mu;
    var += dx * dx + dy * dy + dz * dz + dw * dw;
  }
  var *= (1.f / 96.f);
  float inv = rsqrtf(var + 1e-5f);
  float4* o = (float4*)(out + (size_t)i * 96);
#pragma unroll
  for (int c = 0; c < 24; ++c) {
    float4 v;
    v.x = (buf[c].x - mu) * inv;
    v.y = (buf[c].y - mu) * inv;
    v.z = (buf[c].z - mu) * inv;
    v.w = (buf[c].w - mu) * inv;
    o[c] = v;
  }
}

// ---------------- host ----------------
static inline int cdiv(long long a, long long b) { return (int)((a + b - 1) / b); }

extern "C" void kernel_launch(void* const* d_in, const int* in_sizes, int n_in,
                              void* d_out, int out_size, void* d_ws,
                              size_t ws_size, hipStream_t stream) {
  const float* x_in = (const float*)d_in[0];
  const long long* eidx = (const long long*)d_in[1];
  const float* edge_attr = (const float*)d_in[2];
  const float* W_enter = (const float*)d_in[3];
  const float* b_enter = (const float*)d_in[4];
  const float* e_lin1 = (const float*)d_in[5];
  const float* e_lin2 = (const float*)d_in[6];
  const float* e_att_l = (const float*)d_in[7];
  const float* e_att_r = (const float*)d_in[8];
  const float* e_bias = (const float*)d_in[9];
  const float* gat_W = (const float*)d_in[10];
  const float* gat_att_src = (const float*)d_in[11];
  const float* gat_att_dst = (const float*)d_in[12];
  const float* gat_bias = (const float*)d_in[13];
  const float* gru_Wih = (const float*)d_in[14];
  const float* gru_Whh = (const float*)d_in[15];
  const float* gru_bih = (const float*)d_in[16];
  const float* gru_bhh = (const float*)d_in[17];

  const int N = in_sizes[0] / 64;
  const int E = in_sizes[2] / 16;
  const long long* src = eidx;
  const long long* dstp = eidx + E;

  // ---- workspace carve-out ----
  char* w = (char*)d_ws;
  size_t off = 0;
  auto alloc = [&](size_t bytes) {
    void* p = w + off;
    off = (off + bytes + 255) & ~(size_t)255;
    return p;
  };
  float* x32 = (float*)alloc((size_t)N * 96 * 4);
  _Float16* x16 = (_Float16*)alloc((size_t)N * 96 * 2);
  _Float16* xin16 = (_Float16*)alloc((size_t)N * 64 * 2);
  float* h32 = (float*)alloc((size_t)N * 96 * 4);
  _Float16* h16 = (_Float16*)alloc((size_t)N * 96 * 2);
  float* gi = (float*)alloc((size_t)N * 288 * 4);
  float* gh = (float*)alloc((size_t)N * 288 * 4);
  float* xw = (float*)alloc((size_t)N * 96 * 4);
  float* a_src = (float*)alloc((size_t)N * 4);
  float* a_dst = (float*)alloc((size_t)N * 4);
  float* s_sum = (float*)alloc((size_t)N * 4);
  float* e_edge = (float*)alloc((size_t)E * 4);
  _Float16* ea16 = (_Float16*)alloc((size_t)E * 16 * 2);
  _Float16* pWenter = (_Float16*)alloc((size_t)2 * 6 * 512 * 2);   // K=64
  _Float16* pE1 = (_Float16*)alloc((size_t)4 * 6 * 512 * 2);       // Kpad=128
  _Float16* pE2 = (_Float16*)alloc((size_t)3 * 6 * 512 * 2);       // K=96
  _Float16* pGat[2];
  for (int l = 0; l < 2; ++l) pGat[l] = (_Float16*)alloc((size_t)3 * 6 * 512 * 2);
  _Float16* pWih[3];
  _Float16* pWhh[3];
  for (int l = 0; l < 3; ++l) pWih[l] = (_Float16*)alloc((size_t)3 * 18 * 512 * 2);
  for (int l = 0; l < 3; ++l) pWhh[l] = (_Float16*)alloc((size_t)3 * 18 * 512 * 2);

  const int B = 256;
  // ---- pack weights (f32 -> f16 WMMA-B swizzle) ----
  auto pack = [&](const float* W_, _Float16* out, int K, int Nc, int Kpad,
                  int trans) {
    int total = (Kpad / 32) * (Nc / 16) * 512;
    k_pack<<<cdiv(total, B), B, 0, stream>>>(W_, out, K, Nc, Kpad, trans);
  };
  pack(W_enter, pWenter, 64, 96, 64, 0);
  pack(e_lin1, pE1, 112, 96, 128, 0);
  pack(e_lin2, pE2, 96, 96, 96, 0);
  for (int l = 0; l < 2; ++l) pack(gat_W + (size_t)l * 96 * 96, pGat[l], 96, 96, 96, 0);
  for (int l = 0; l < 3; ++l) {
    pack(gru_Wih + (size_t)l * 288 * 96, pWih[l], 96, 288, 96, 1);  // B = Wih^T
    pack(gru_Whh + (size_t)l * 288 * 96, pWhh[l], 96, 288, 96, 1);
  }
  k_cvt<<<cdiv((long long)N * 64, B), B, 0, stream>>>(x_in, xin16, (size_t)N * 64);
  k_cvt<<<cdiv((long long)E * 16, B), B, 0, stream>>>(edge_attr, ea16, (size_t)E * 16);

  // one wave = 16 rows x 96 cols (6 tiles), A fragment loaded once per K-chunk
  auto gemm = [&](const _Float16* A, int lda, const _Float16* Bp, int K, int Nc,
                  const float* bias, int act, float* C, _Float16* C16, int M) {
    long long waves = (long long)((M + 15) / 16) * (Nc / 96);
    k_gemm<6><<<cdiv(waves, 8), B, 0, stream>>>(A, lda, Bp, K, Nc, bias, act, C,
                                                C16, M);
  };
  auto run_gru = [&](int l) {
    gemm(h16, 96, pWih[l], 96, 288, gru_bih + (size_t)l * 288, 0, gi, nullptr, N);
    gemm(x16, 96, pWhh[l], 96, 288, gru_bhh + (size_t)l * 288, 0, gh, nullptr, N);
    k_gru_combine<<<cdiv((long long)N * 96, B), B, 0, stream>>>(gi, gh, x32, x16,
                                                                (size_t)N * 96);
  };

  // ---- entry linear + leaky ----
  gemm(xin16, 64, pWenter, 64, 96, b_enter, 1, x32, x16, N);

  // ---- Edge2dConv layer 0 ----
  k_zero<<<cdiv(N, B), B, 0, stream>>>(s_sum, (size_t)N);
  k_zero<<<cdiv((long long)N * 96, B), B, 0, stream>>>(h32, (size_t)N * 96);
  int etiles = (E + 15) / 16;
  k_edge_logits<<<cdiv(etiles, 8), B, 0, stream>>>(x16, ea16, src, dstp, pE1,
                                                   e_att_l, e_att_r, x32, e_edge,
                                                   s_sum, E);
  k_edge_msg<<<cdiv(etiles, 8), B, 0, stream>>>(x16, ea16, src, dstp, pE1, pE2,
                                                e_edge, s_sum, h32, E);
  k_elu_bias<<<cdiv((long long)N * 96, B), B, 0, stream>>>(h32, e_bias, h16,
                                                           (size_t)N * 96);
  run_gru(0);

  // ---- GAT layers ----
  for (int l = 0; l < 2; ++l) {
    gemm(x16, 96, pGat[l], 96, 96, nullptr, 0, xw, nullptr, N);
    k_node_att<<<cdiv(N, B), B, 0, stream>>>(xw, gat_att_src + (size_t)l * 96,
                                             gat_att_dst + (size_t)l * 96, a_src,
                                             a_dst, N);
    k_zero<<<cdiv(N, B), B, 0, stream>>>(s_sum, (size_t)N);
    k_zero<<<cdiv((long long)N * 96, B), B, 0, stream>>>(h32, (size_t)N * 96);
    k_gat_logits<<<cdiv(E, B), B, 0, stream>>>(a_src, a_dst, src, dstp, e_edge,
                                               s_sum, E);
    k_gat_scatter<<<cdiv((long long)E * 24, B), B, 0, stream>>>(xw, src, dstp,
                                                                e_edge, s_sum,
                                                                h32, E);
    k_elu_bias<<<cdiv((long long)N * 96, B), B, 0, stream>>>(
        h32, gat_bias + (size_t)l * 96, h16, (size_t)N * 96);
    run_gru(l + 1);
  }

  // ---- final layer norm ----
  k_layernorm<<<cdiv(N, B), B, 0, stream>>>(x32, (float*)d_out, N);
}
